// TripletGraphTransformerLayer_87187836109387
// MI455X (gfx1250) — matrix operand order repaired and checked by reading, not compiled, not run on hardware
//
#include <hip/hip_runtime.h>
#include <hip/hip_bf16.h>
#include <math.h>

// ---------------------------------------------------------------------------
// CDNA5 (gfx1250) bf16 WMMA types & helpers
// ---------------------------------------------------------------------------
typedef __bf16 bf16;
typedef __attribute__((ext_vector_type(16))) __bf16 v16bf;
typedef __attribute__((ext_vector_type(8)))  __bf16 v8bf;
typedef __attribute__((ext_vector_type(4)))  __bf16 v4bf;
typedef __attribute__((ext_vector_type(8)))  float  v8f;

#define N_   256
#define DN_  256
#define DE_  128
#define H_   8
#define HT_  4
#define DT_  32
#define DHN_ 32
#define EPS_ 1e-5f
#define INV_SQRT32 0.1767766952966369f

__device__ inline v8f v8f_zero() {
  v8f z;
#pragma unroll
  for (int i = 0; i < 8; ++i) z[i] = 0.f;
  return z;
}

__device__ inline v8f wmma_bf16(v16bf a, v16bf b, v8f c) {
  return __builtin_amdgcn_wmma_f32_16x16x32_bf16(false, a, false, b,
                                                 (short)0, c, false, false);
}

// A fragment from row-major bf16 LDS tile (ld = row stride, elements).
// lane m = L&15 ; halves t<8 -> K = hi*8+t ; t>=8 -> K = 16+hi*8+(t-8).
__device__ inline v16bf frag_a(const bf16* buf, int ld, int r0, int k0, int lane) {
  const bf16* row = buf + (size_t)(r0 + (lane & 15)) * ld + k0 + (lane >> 4) * 8;
  v8bf lo = *(const v8bf*)(row);
  v8bf hi = *(const v8bf*)(row + 16);
  return __builtin_shufflevector(lo, hi, 0, 1, 2, 3, 4, 5, 6, 7,
                                 8, 9, 10, 11, 12, 13, 14, 15);
}

// B fragment from TRANSPOSED LDS tile T[n][k] (ld = k-extent row stride).
// lane n = L&15 ; t<8 -> K = (L>>4)*8+t ; t>=8 -> K = 16+(L>>4)*8+(t-8).
__device__ inline v16bf frag_bT(const bf16* buf, int ld, int k0, int n0, int lane) {
  const bf16* row = buf + (size_t)(n0 + (lane & 15)) * ld + k0 + (lane >> 4) * 8;
  v8bf lo = *(const v8bf*)(row);
  v8bf hi = *(const v8bf*)(row + 16);
  return __builtin_shufflevector(lo, hi, 0, 1, 2, 3, 4, 5, 6, 7,
                                 8, 9, 10, 11, 12, 13, 14, 15);
}

// convert 8 consecutive f32 (32B-aligned) -> v8bf
__device__ inline v8bf cvt8(const float* p) {
  float4 f0 = *(const float4*)(p);
  float4 f1 = *(const float4*)(p + 4);
  v8bf o;
  o[0] = (bf16)f0.x; o[1] = (bf16)f0.y; o[2] = (bf16)f0.z; o[3] = (bf16)f0.w;
  o[4] = (bf16)f1.x; o[5] = (bf16)f1.y; o[6] = (bf16)f1.z; o[7] = (bf16)f1.w;
  return o;
}

// Transposed store of a 4(k) x 4(n) micro-tile: src rows are 4 coalesced
// float4 loads; dst gets one packed v4bf (ds_store_b64) per n.
// dstT layout: [n][ldk], k contiguous. r0 must be a multiple of 4.
__device__ inline void tstore4x4(const float* __restrict__ src, size_t srcStride,
                                 bf16* dstT, int ldk, int r0, int c0) {
  float fv[4][4];
#pragma unroll
  for (int rr = 0; rr < 4; ++rr) {
    float4 t = *(const float4*)(src + (size_t)rr * srcStride);
    fv[rr][0] = t.x; fv[rr][1] = t.y; fv[rr][2] = t.z; fv[rr][3] = t.w;
  }
#pragma unroll
  for (int cc = 0; cc < 4; ++cc) {
    v4bf p;
#pragma unroll
    for (int rr = 0; rr < 4; ++rr) p[rr] = (bf16)fv[rr][cc];
    *(v4bf*)(dstT + (size_t)(c0 + cc) * ldk + r0) = p;
  }
}

__device__ inline float gelu_exact(float x) {
  return 0.5f * x * (1.f + erff(x * 0.70710678118654752f));
}
__device__ inline float sigmoidf_(float x) { return 1.f / (1.f + __expf(-x)); }

// ---------------------------------------------------------------------------
// Generic GEMM: C = act(A[MxK] @ B[KxN] + bias) + resScale*res  (bf16 WMMA)
// Tiles 64x128x32, 256 threads = 8 wave32 (4x2 waves, 4 accumulators each).
// ---------------------------------------------------------------------------
__global__ __launch_bounds__(256) void gemm_k(
    const float* __restrict__ A, const float* __restrict__ B,
    const float* __restrict__ bias, const float* __restrict__ res,
    float resScale, float* __restrict__ C, int M, int Nn, int K, int dogelu) {
  __shared__ __attribute__((aligned(16))) bf16 As[64 * 32];    //  4 KB
  __shared__ __attribute__((aligned(16))) bf16 BsT[128 * 32];  //  8 KB
  int tid = threadIdx.x, lane = tid & 31, w = tid >> 5;
  int wr = w >> 1, wc = w & 1;
  int m0 = blockIdx.y * 64, n0 = blockIdx.x * 128;

  v8f acc[4];
#pragma unroll
  for (int s = 0; s < 4; ++s) acc[s] = v8f_zero();

  for (int kk = 0; kk < K; kk += 32) {
    __syncthreads();
    { // A tile 64x32, 8 elems/thread, vectorized both sides
      int i0 = tid * 8, r = i0 >> 5, c = i0 & 31;
      *(v8bf*)(As + i0) = cvt8(A + (size_t)(m0 + r) * K + kk + c);
    }
    { // B tile 32x128 -> BsT[128][32] via packed 4x4 transpose stores
      int c0 = (tid & 31) * 4;   // n within tile (quad)
      int r0 = (tid >> 5) * 4;   // k within tile (quad)
      tstore4x4(B + (size_t)(kk + r0) * Nn + n0 + c0, Nn, BsT, 32, r0, c0);
    }
    __syncthreads();
    v16bf a = frag_a(As, 32, wr * 16, 0, lane);
#pragma unroll
    for (int s = 0; s < 4; ++s) {
      v16bf b = frag_bT(BsT, 32, 0, wc * 64 + s * 16, lane);
      acc[s] = wmma_bf16(a, b, acc[s]);
    }
  }
  int rowb = m0 + wr * 16 + (lane >> 4) * 8;
#pragma unroll
  for (int s = 0; s < 4; ++s) {
    int col = n0 + wc * 64 + s * 16 + (lane & 15);
    float bb = bias ? bias[col] : 0.f;
    if (res) {
#pragma unroll
      for (int r = 0; r < 8; ++r) {
        float x = acc[s][r] + bb;
        if (dogelu) x = gelu_exact(x);
        C[(size_t)(rowb + r) * Nn + col] =
            x + resScale * res[(size_t)(rowb + r) * Nn + col];
      }
    } else {
#pragma unroll
      for (int r = 0; r < 8; ++r) {
        float x = acc[s][r] + bb;
        if (dogelu) x = gelu_exact(x);
        C[(size_t)(rowb + r) * Nn + col] = x;
      }
    }
  }
}

// ---------------------------------------------------------------------------
// Small-N projection: Y[row,c] = X[row,:K] @ W[K,C] (+bias), C<=32, block==K==128
// ---------------------------------------------------------------------------
__global__ void colsmall_k(const float* __restrict__ X, const float* __restrict__ W,
                           const float* __restrict__ bias, float* __restrict__ Y,
                           int K, int C) {
  int row = blockIdx.x, t = threadIdx.x;
  __shared__ float xr[128];
  xr[t] = X[(size_t)row * K + t];
  __syncthreads();
  if (t < C) {
    float acc = bias ? bias[t] : 0.f;
    for (int k = 0; k < K; ++k) acc += xr[k] * W[(size_t)k * C + t];
    Y[(size_t)row * C + t] = acc;
  }
}

// ---------------------------------------------------------------------------
// Node attention: raw = qk*scale + bias ; attn = softmax(raw*gate) ; o = attn@v
// grid = (N, H), block = 256 (thread = j)
// ---------------------------------------------------------------------------
__global__ __launch_bounds__(256) void node_attn_k(
    const float* __restrict__ q, const float* __restrict__ k,
    const float* __restrict__ v, const float* __restrict__ bg,
    float* __restrict__ raw, float* __restrict__ o) {
  int i = blockIdx.x, hh = blockIdx.y, j = threadIdx.x;
  __shared__ float sw[256], sr[256];
  const float* qi = q + (size_t)i * DN_ + hh * DHN_;
  const float* kj = k + (size_t)j * DN_ + hh * DHN_;
  float dot = 0.f;
#pragma unroll
  for (int d = 0; d < DHN_; ++d) dot += qi[d] * kj[d];
  size_t ij = (size_t)i * N_ + j;
  float rawv = dot * INV_SQRT32 + bg[ij * 16 + hh];
  raw[ij * 8 + hh] = rawv;
  float gt = sigmoidf_(bg[ij * 16 + 8 + hh]);
  float sc = rawv * gt;
  sr[j] = sc;
  __syncthreads();
  for (int s = 128; s > 0; s >>= 1) {
    if (j < s) sr[j] = fmaxf(sr[j], sr[j + s]);
    __syncthreads();
  }
  float m = sr[0];
  __syncthreads();
  float wv = __expf(sc - m);
  sw[j] = wv; sr[j] = wv;
  __syncthreads();
  for (int s = 128; s > 0; s >>= 1) {
    if (j < s) sr[j] += sr[j + s];
    __syncthreads();
  }
  float inv = 1.f / sr[0];
  if (j < DHN_) {
    float acc = 0.f;
    for (int jj = 0; jj < N_; ++jj)
      acc += sw[jj] * v[(size_t)jj * DN_ + hh * DHN_ + j];
    o[(size_t)i * DN_ + hh * DHN_ + j] = acc * inv;
  }
}

// ---------------------------------------------------------------------------
// Row LayerNorm (blockDim.x == C, C in {128,256})
// ---------------------------------------------------------------------------
__global__ void ln_k(const float* __restrict__ x, float* __restrict__ y, int C) {
  int row = blockIdx.x, t = threadIdx.x;
  __shared__ float s1[256], s2[256];
  float v = x[(size_t)row * C + t];
  s1[t] = v; s2[t] = v * v;
  __syncthreads();
  for (int s = blockDim.x >> 1; s > 0; s >>= 1) {
    if (t < s) { s1[t] += s1[t + s]; s2[t] += s2[t + s]; }
    __syncthreads();
  }
  float mean = s1[0] / C;
  float var = s2[0] / C - mean * mean;
  y[(size_t)row * C + t] = (v - mean) * rsqrtf(var + EPS_);
}

__global__ void add_k(const float* a, const float* b, float* y, int n) {
  int i = blockIdx.x * blockDim.x + threadIdx.x;
  if (i < n) y[i] = a[i] + b[i];
}

// ---------------------------------------------------------------------------
// e_agg = ln(e)[i,j,:] + raw[i,j,:8] @ Wl2e + bl2e    (block = 128)
// ---------------------------------------------------------------------------
__global__ void edge_agg_k(const float* __restrict__ e, const float* __restrict__ raw,
                           const float* __restrict__ Wl2e, const float* __restrict__ bl2e,
                           float* __restrict__ out) {
  int row = blockIdx.x, t = threadIdx.x;
  __shared__ float s1[128], s2[128], rawv[8];
  float v = e[(size_t)row * DE_ + t];
  s1[t] = v; s2[t] = v * v;
  if (t < 8) rawv[t] = raw[(size_t)row * 8 + t];
  __syncthreads();
  for (int s = 64; s > 0; s >>= 1) {
    if (t < s) { s1[t] += s1[t + s]; s2[t] += s2[t + s]; }
    __syncthreads();
  }
  float mean = s1[0] / DE_, var = s2[0] / DE_ - mean * mean;
  float lnv = (v - mean) * rsqrtf(var + EPS_);
  float acc = bl2e[t];
#pragma unroll
  for (int tt = 0; tt < 8; ++tt) acc += rawv[tt] * Wl2e[tt * DE_ + t];
  out[(size_t)row * DE_ + t] = lnv + acc;
}

// ---------------------------------------------------------------------------
// Fused edge FFN: out = resScale*res + gelu(X@W1+b1)@W2+b2
// X = pre-LayerNorm'd input [rows x 128]; 16 rows/workgroup, 8 waves.
// Hidden (16x512) lives entirely in LDS as bf16 — never touches HBM.
// ---------------------------------------------------------------------------
__global__ __launch_bounds__(256) void ffn_fused_k(
    const float* __restrict__ X, const float* __restrict__ W1,
    const float* __restrict__ b1, const float* __restrict__ W2,
    const float* __restrict__ b2, const float* __restrict__ res,
    float resScale, float* __restrict__ out) {
  __shared__ __attribute__((aligned(16))) bf16 Xs[16 * 128];   //  4 KB
  __shared__ __attribute__((aligned(16))) bf16 WsT[512 * 32];  // 32 KB (W1 slab; reused for W2)
  __shared__ __attribute__((aligned(16))) bf16 Hs[16 * 512];   // 16 KB
  int tid = threadIdx.x, lane = tid & 31, w = tid >> 5;
  int r0 = blockIdx.x * 16;

  { // X tile 16x128, vectorized
    int i0 = tid * 8, r = i0 >> 7, c = i0 & 127;
    *(v8bf*)(Xs + i0) = cvt8(X + (size_t)(r0 + r) * 128 + c);
  }

  // stage 1: hidden = gelu(X @ W1 + b1); wave w owns hidden cols [w*64, w*64+64)
  v8f acc[4];
#pragma unroll
  for (int s = 0; s < 4; ++s) acc[s] = v8f_zero();
  for (int kk = 0; kk < 128; kk += 32) {
    __syncthreads();
#pragma unroll
    for (int u = 0; u < 4; ++u) { // W1 slab 32x512 -> WsT[512][32] (4x4 packed)
      int idx = tid + u * 256;
      int c0 = (idx & 127) * 4;  // n quad (coalesced fastest)
      int kq = (idx >> 7) * 4;   // k quad
      tstore4x4(W1 + (size_t)(kk + kq) * 512 + c0, 512, WsT, 32, kq, c0);
    }
    __syncthreads();
    v16bf a = frag_a(Xs, 128, 0, kk, lane);
#pragma unroll
    for (int s = 0; s < 4; ++s) {
      v16bf b = frag_bT(WsT, 32, 0, w * 64 + s * 16, lane);
      acc[s] = wmma_bf16(a, b, acc[s]);
    }
  }
  __syncthreads();
#pragma unroll
  for (int s = 0; s < 4; ++s) {
    int col = w * 64 + s * 16 + (lane & 15);
    int rb = (lane >> 4) * 8;
    float bb = b1[col];
#pragma unroll
    for (int r = 0; r < 8; ++r) {
      float x = acc[s][r] + bb;
      Hs[(size_t)(rb + r) * 512 + col] = (bf16)gelu_exact(x);
    }
  }
  __syncthreads();

  // stage 2: out = hidden @ W2 + b2 + resScale*res; wave w owns cols [w*16, w*16+16)
  v8f acc2 = v8f_zero();
  for (int kk = 0; kk < 512; kk += 32) {
    { // W2 slab 32x128 -> WsT[128][32] (4x4 packed)
      int c0 = (tid & 31) * 4;
      int kq = (tid >> 5) * 4;
      tstore4x4(W2 + (size_t)(kk + kq) * 128 + c0, 128, WsT, 32, kq, c0);
    }
    __syncthreads();
    v16bf a = frag_a(Hs, 512, 0, kk, lane);
    v16bf b = frag_bT(WsT, 32, 0, w * 16, lane);
    acc2 = wmma_bf16(a, b, acc2);
    __syncthreads();
  }
  int col = w * 16 + (lane & 15);
  int rb = (lane >> 4) * 8;
  float bb = b2[col];
#pragma unroll
  for (int r = 0; r < 8; ++r) {
    int row = r0 + rb + r;
    out[(size_t)row * 128 + col] =
        acc2[r] + bb + resScale * res[(size_t)row * 128 + col];
  }
}

// ---------------------------------------------------------------------------
// Fused triplet attention. One workgroup per (j, h). Exploits
// dot_out[h,i,k,j] == dot_in[h,i,j,k]: the score tile is computed ONCE and
// reused by both softmax directions. Per (h,j): S = Q_j(256x32) @ P_j^T ;
// O_{in,out}(16x32) = A(16x256) @ V_j(256x32).
// ---------------------------------------------------------------------------
__global__ __launch_bounds__(256) void triplet_k(
    const float* __restrict__ qt, const float* __restrict__ pt,
    const float* __restrict__ vt, const float* __restrict__ gbt,
    float* __restrict__ opair) {
  __shared__ __attribute__((aligned(16))) bf16 Ps[256 * 32];   // 16 KB  [k][d]
  __shared__ __attribute__((aligned(16))) bf16 VsT[32 * 256];  // 16 KB  [d][k]
  __shared__ __attribute__((aligned(16))) bf16 Qs[16 * 32];    //  1 KB
  __shared__ __attribute__((aligned(16))) float Ss[16 * 256];  // 16 KB
  __shared__ __attribute__((aligned(16))) bf16 Aw[16 * 256];   //  8 KB
  __shared__ float gateO[256], biasO[256];
  __shared__ float red[16][17];
  __shared__ float invsum[16];

  int j = blockIdx.x, hh = blockIdx.y;
  int tid = threadIdx.x, lane = tid & 31, w = tid >> 5;
  int row = tid >> 4, sub = tid & 15;

#pragma unroll
  for (int u = 0; u < 4; ++u) { // P tile [k][d], contiguous v8bf stores
    int i0 = (tid + u * 256) * 8, kk = i0 >> 5, d = i0 & 31;
    *(v8bf*)(Ps + i0) = cvt8(pt + ((size_t)j * N_ + kk) * DE_ + hh * DT_ + d);
  }
#pragma unroll
  for (int u = 0; u < 2; ++u) { // V -> VsT[32d][256k] (4x4 packed transpose)
    int idx = tid + u * 256;
    int d0 = (idx & 7) * 4;
    int kq = (idx >> 3) * 4;
    tstore4x4(vt + ((size_t)j * N_ + kq) * DE_ + hh * DT_ + d0, DE_,
              VsT, 256, kq, d0);
  }
  { // per-row out-direction gate/bias: gate[h,j,i] / bias[h,j,i]
    int i = tid;
    size_t g = ((size_t)j * N_ + i) * (2 * HT_);
    gateO[i] = sigmoidf_(gbt[g + hh]);
    biasO[i] = gbt[g + HT_ + hh];
  }

  for (int it = 0; it < 16; ++it) {
    int gi0 = it * 16;
    __syncthreads();
    if (tid < 64) { // Q tile 16x32
      int i0 = tid * 8, r = i0 >> 5, d = i0 & 31;
      *(v8bf*)(Qs + i0) = cvt8(qt + ((size_t)(gi0 + r) * N_ + j) * DE_ + hh * DT_ + d);
    }
    __syncthreads();

    // S(16x256) = Q(16x32) @ P^T (Ps IS the transposed-B layout), 8 waves x 2 tiles
    {
      v16bf a = frag_a(Qs, 32, 0, 0, lane);
#pragma unroll
      for (int s = 0; s < 2; ++s) {
        int n0 = w * 32 + s * 16;
        v16bf b = frag_bT(Ps, 32, 0, n0, lane);
        v8f c = wmma_bf16(a, b, v8f_zero());
        int colD = n0 + (lane & 15);
        int rbD = (lane >> 4) * 8;
#pragma unroll
        for (int r = 0; r < 8; ++r) Ss[(size_t)(rbD + r) * 256 + colD] = c[r] * INV_SQRT32;
      }
    }
    __syncthreads();

    // ---- out-direction softmax over k (gate/bias constant per row) ----
    {
      int gi = gi0 + row;
      float gt = gateO[gi], bs = biasO[gi];
      float gv[16];
      float mx = -3.4e38f;
#pragma unroll
      for (int t = 0; t < 16; ++t) {
        float g = (Ss[(size_t)row * 256 + sub * 16 + t] + bs) * gt;
        gv[t] = g; mx = fmaxf(mx, g);
      }
      red[row][sub] = mx;
      __syncthreads();
      if (sub == 0) {
        float m = red[row][0];
        for (int t = 1; t < 16; ++t) m = fmaxf(m, red[row][t]);
        red[row][16] = m;
      }
      __syncthreads();
      float m = red[row][16], sm = 0.f;
#pragma unroll
      for (int t = 0; t < 16; ++t) {
        float wv = __expf(gv[t] - m);
        Aw[(size_t)row * 256 + sub * 16 + t] = (bf16)wv;
        sm += wv;
      }
      red[row][sub] = sm;
      __syncthreads();
      if (sub == 0) {
        float s = 0.f;
        for (int t = 0; t < 16; ++t) s += red[row][t];
        invsum[row] = 1.f / s;
      }
    }
    __syncthreads();

    // O_out(16x32) = Aw @ V (waves 0,1), normalize on store
    if (w < 2) {
      int n0 = w * 16;
      v8f c = v8f_zero();
#pragma unroll
      for (int kb = 0; kb < 256; kb += 32) {
        v16bf a = frag_a(Aw, 256, 0, kb, lane);
        v16bf b = frag_bT(VsT, 256, kb, n0, lane);
        c = wmma_bf16(a, b, c);
      }
      int d = n0 + (lane & 15);
      int rbD = (lane >> 4) * 8;
#pragma unroll
      for (int r = 0; r < 8; ++r) {
        int m = rbD + r, gi = gi0 + m;
        opair[((size_t)gi * N_ + j) * (2 * DE_) + DE_ + hh * DT_ + d] = c[r] * invsum[m];
      }
    }
    __syncthreads();

    // ---- in-direction softmax over k (gate/bias vary with (i,k)) ----
    {
      int gi = gi0 + row;
      const float* gb = gbt + ((size_t)gi * N_ + sub * 16) * (2 * HT_);
      float gv[16];
      float mx = -3.4e38f;
#pragma unroll
      for (int t = 0; t < 16; ++t) {
        float gt = sigmoidf_(gb[t * (2 * HT_) + hh]);
        float bs = gb[t * (2 * HT_) + HT_ + hh];
        float g = (Ss[(size_t)row * 256 + sub * 16 + t] + bs) * gt;
        gv[t] = g; mx = fmaxf(mx, g);
      }
      red[row][sub] = mx;
      __syncthreads();
      if (sub == 0) {
        float m = red[row][0];
        for (int t = 1; t < 16; ++t) m = fmaxf(m, red[row][t]);
        red[row][16] = m;
      }
      __syncthreads();
      float m = red[row][16], sm = 0.f;
#pragma unroll
      for (int t = 0; t < 16; ++t) {
        float wv = __expf(gv[t] - m);
        Aw[(size_t)row * 256 + sub * 16 + t] = (bf16)wv;
        sm += wv;
      }
      red[row][sub] = sm;
      __syncthreads();
      if (sub == 0) {
        float s = 0.f;
        for (int t = 0; t < 16; ++t) s += red[row][t];
        invsum[row] = 1.f / s;
      }
    }
    __syncthreads();

    // O_in(16x32) = Aw @ V (waves 0,1)
    if (w < 2) {
      int n0 = w * 16;
      v8f c = v8f_zero();
#pragma unroll
      for (int kb = 0; kb < 256; kb += 32) {
        v16bf a = frag_a(Aw, 256, 0, kb, lane);
        v16bf b = frag_bT(VsT, 256, kb, n0, lane);
        c = wmma_bf16(a, b, c);
      }
      int d = n0 + (lane & 15);
      int rbD = (lane >> 4) * 8;
#pragma unroll
      for (int r = 0; r < 8; ++r) {
        int m = rbD + r, gi = gi0 + m;
        opair[((size_t)gi * N_ + j) * (2 * DE_) + hh * DT_ + d] = c[r] * invsum[m];
      }
    }
  }
}

// ---------------------------------------------------------------------------
// Host-side orchestration
// ---------------------------------------------------------------------------
extern "C" void kernel_launch(void* const* d_in, const int* in_sizes, int n_in,
                              void* d_out, int out_size, void* d_ws, size_t ws_size,
                              hipStream_t stream) {
  const float* h_in  = (const float*)d_in[0];
  const float* e_in  = (const float*)d_in[1];
  const float* Wq    = (const float*)d_in[2];
  const float* Wk    = (const float*)d_in[3];
  const float* Wv    = (const float*)d_in[4];
  const float* W_ebg = (const float*)d_in[5];
  const float* Wo    = (const float*)d_in[6];
  const float* bo    = (const float*)d_in[7];
  const float* Wn1   = (const float*)d_in[8];
  const float* bn1   = (const float*)d_in[9];
  const float* Wn2   = (const float*)d_in[10];
  const float* bn2   = (const float*)d_in[11];
  const float* We1   = (const float*)d_in[12];
  const float* be1   = (const float*)d_in[13];
  const float* We2   = (const float*)d_in[14];
  const float* be2   = (const float*)d_in[15];
  const float* Wl2e  = (const float*)d_in[16];
  const float* bl2e  = (const float*)d_in[17];
  const float* Wqt   = (const float*)d_in[18];
  const float* Wpt   = (const float*)d_in[19];
  const float* Wvt   = (const float*)d_in[20];
  const float* Wgbt  = (const float*)d_in[21];
  const float* Wto   = (const float*)d_in[22];
  const float* bto   = (const float*)d_in[23];

  float* hout = (float*)d_out;              // [256,256]
  float* eout = hout + 256 * 256;           // [256,256,128]

  // workspace layout (floats)
  const size_t EC = (size_t)65536 * 128;
  float* W = (float*)d_ws;
  float* qn  = W + 0;
  float* kn  = W + 65536;
  float* vn  = W + 131072;
  float* on_ = W + 196608;
  float* h1  = W + 262144;
  float* hn  = W + 327680;
  float* rr  = W + 393216;
  float* t2  = W + 458752;
  float* u2  = W + 524288;       // 256*1024
  float* raw = W + 786432;       // 256*256*8
  float* bg  = W + 1310720;      // 256*256*16
  float* gbt = W + 2359296;      // 256*256*8
  float* base = W + 2883584;
  float* eagg  = base;           // also reused as e3
  float* lnb   = base + 1 * EC;
  float* e2b   = base + 2 * EC;
  float* qtb   = base + 3 * EC;
  float* ptb   = base + 4 * EC;
  float* vtb   = base + 5 * EC;
  float* opair = base + 6 * EC;  // 2*EC
  float* e3b   = eagg;

  // ---- node attention ----
  gemm_k<<<dim3(2, 4), 256, 0, stream>>>(h_in, Wq, nullptr, nullptr, 0.f, qn, 256, 256, 256, 0);
  gemm_k<<<dim3(2, 4), 256, 0, stream>>>(h_in, Wk, nullptr, nullptr, 0.f, kn, 256, 256, 256, 0);
  gemm_k<<<dim3(2, 4), 256, 0, stream>>>(h_in, Wv, nullptr, nullptr, 0.f, vn, 256, 256, 256, 0);
  colsmall_k<<<65536, 128, 0, stream>>>(e_in, W_ebg, nullptr, bg, 128, 16);
  node_attn_k<<<dim3(256, 8), 256, 0, stream>>>(qn, kn, vn, bg, raw, on_);
  gemm_k<<<dim3(2, 4), 256, 0, stream>>>(on_, Wo, bo, h_in, 1.f, h1, 256, 256, 256, 0);

  // ---- edge aggregation + first FFN block: e2 = 2*e_agg + ffn(ln(e_agg)) ----
  edge_agg_k<<<65536, 128, 0, stream>>>(e_in, raw, Wl2e, bl2e, eagg);
  ln_k<<<65536, 128, 0, stream>>>(eagg, lnb, 128);
  ffn_fused_k<<<4096, 256, 0, stream>>>(lnb, We1, be1, We2, be2, eagg, 2.f, e2b);

  // ---- triplet module on ln(e2) ----
  ln_k<<<65536, 128, 0, stream>>>(e2b, lnb, 128);
  gemm_k<<<dim3(1, 1024), 256, 0, stream>>>(lnb, Wqt, nullptr, nullptr, 0.f, qtb, 65536, 128, 128, 0);
  gemm_k<<<dim3(1, 1024), 256, 0, stream>>>(lnb, Wpt, nullptr, nullptr, 0.f, ptb, 65536, 128, 128, 0);
  gemm_k<<<dim3(1, 1024), 256, 0, stream>>>(lnb, Wvt, nullptr, nullptr, 0.f, vtb, 65536, 128, 128, 0);
  colsmall_k<<<65536, 128, 0, stream>>>(lnb, Wgbt, nullptr, gbt, 128, 8);
  triplet_k<<<dim3(256, 4), 256, 0, stream>>>(qtb, ptb, vtb, gbt, opair);
  gemm_k<<<dim3(1, 1024), 256, 0, stream>>>(opair, Wto, bto, e2b, 1.f, e3b, 65536, 128, 256, 0);

  // ---- second FFN block: e4 = 2*e3 + ffn(ln(e3)) -> output ----
  ln_k<<<65536, 128, 0, stream>>>(e3b, lnb, 128);
  ffn_fused_k<<<4096, 256, 0, stream>>>(lnb, We1, be1, We2, be2, e3b, 2.f, eout);

  // ---- node FFN: h_out = h1 + ln(h1) + ffn_n(ln(ln(h1))) ----
  ln_k<<<256, 256, 0, stream>>>(h1, hn, 256);
  add_k<<<256, 256, 0, stream>>>(h1, hn, rr, 65536);
  ln_k<<<256, 256, 0, stream>>>(hn, t2, 256);
  gemm_k<<<dim3(8, 4), 256, 0, stream>>>(t2, Wn1, bn1, nullptr, 0.f, u2, 256, 1024, 256, 1);
  gemm_k<<<dim3(2, 4), 256, 0, stream>>>(u2, Wn2, bn2, rr, 1.f, hout, 256, 256, 1024, 0);
}
